// Sage_En_18940805775915
// MI455X (gfx1250) — matrix operand order repaired and checked by reading, not compiled, not run on hardware
//
#include <hip/hip_runtime.h>
#include <hip/hip_bf16.h>

typedef _Float16 v16h __attribute__((ext_vector_type(16)));
typedef _Float16 v8h  __attribute__((ext_vector_type(8)));
typedef float    v8f  __attribute__((ext_vector_type(8)));
typedef unsigned int v4u __attribute__((ext_vector_type(4)));
typedef int      v4i __attribute__((ext_vector_type(4)));
typedef int      v8i __attribute__((ext_vector_type(8)));

#define NPTS   16384
#define NFEAT  256
#define NEMBED 128
#define HID1   64
#define HID2   32
#define NOUT   8
#define XNLD   512   // XN row stride in halves: [x f16 (256) | neigh f16 (256)]

// ---- WMMA wrapper (device-only builtin) ----
__device__ __forceinline__ v8f wmma_f16(v16h a, v16h b, v8f c) {
#if defined(__HIP_DEVICE_COMPILE__)
  return __builtin_amdgcn_wmma_f32_16x16x32_f16(false, a, false, b, (short)0, c, false, false);
#else
  return c;
#endif
}

__device__ __forceinline__ v16h pack16(v8h lo, v8h hi) {
  union { v16h v; v8h h[2]; } u;
  u.h[0] = lo; u.h[1] = hi;
  return u.v;
}

// ============================================================
// TDM: DMA a 2D f32 tile (tile_d0 elems x tile_d1 rows) from global
// memory (row stride = stride0 elems) into LDS at lds_byte_addr.
// LDS padding: +4 dwords after every 64 dwords (one 64-elem row)
// -> effective LDS row stride = 68 dwords (bank-conflict-free).
// ============================================================
__device__ __forceinline__ void tdm_load_tile_f32(
    unsigned int lds_byte_addr, const float* gptr,
    unsigned int tile_d0, unsigned int tile_d1,
    unsigned int tensor_d0, unsigned int tensor_d1,
    unsigned int stride0)
{
#if defined(__HIP_DEVICE_COMPILE__)
  unsigned long long ga = (unsigned long long)(size_t)gptr;
  v4u g0;
  g0[0] = 1u;                                      // count=1, user mode, no gather
  g0[1] = lds_byte_addr;                           // LDS byte address
  g0[2] = (unsigned int)(ga & 0xffffffffu);        // global_addr[31:0]
  g0[3] = (unsigned int)((ga >> 32) & 0x01ffffffu) // global_addr[56:32]
        | (2u << 30);                              // type=2 (image)
  v8i g1;
  // data_size=2 (4B) | pad_enable | pad_interval=5 (64 dwords) | pad_amount=3 (4 dwords)
  g1[0] = (int)((2u << 16) | (1u << 20) | (5u << 22) | (3u << 25));
  g1[1] = (int)((tensor_d0 & 0xffffu) << 16);                       // tensor_dim0 lo
  g1[2] = (int)((tensor_d0 >> 16) | ((tensor_d1 & 0xffffu) << 16)); // dim0 hi | dim1 lo
  g1[3] = (int)((tensor_d1 >> 16) | ((tile_d0 & 0xffffu) << 16));   // dim1 hi | tile_dim0
  g1[4] = (int)(tile_d1 & 0xffffu);                                 // tile_dim1 (tile_dim2=0)
  g1[5] = (int)stride0;                                             // tensor_dim0_stride[31:0]
  g1[6] = 0;                                                        // stride0 hi | stride1 lo
  g1[7] = 0;                                                        // stride1 hi
  v4i g2; g2[0] = 0; g2[1] = 0; g2[2] = 0; g2[3] = 0;               // 2D: groups 2/3 unused
  v4i g3; g3[0] = 0; g3[1] = 0; g3[2] = 0; g3[3] = 0;
#if defined(__clang_major__) && (__clang_major__ >= 23)
  v8i g4; g4[0]=0; g4[1]=0; g4[2]=0; g4[3]=0; g4[4]=0; g4[5]=0; g4[6]=0; g4[7]=0;
  __builtin_amdgcn_tensor_load_to_lds(g0, g1, g2, g3, g4, 0);
#else
  __builtin_amdgcn_tensor_load_to_lds(g0, g1, g2, g3, 0);
#endif
#endif
}

__device__ __forceinline__ void wait_tensor0() {
#if defined(__HIP_DEVICE_COMPILE__)
  __builtin_amdgcn_s_wait_tensorcnt(0);
#endif
}

// ============================================================
// Kernel 0a: x f32 -> XN[:, 0:256] f16 (row-major, stride 512)
// ============================================================
__global__ void k_prep_x(const float* __restrict__ x, _Float16* __restrict__ xn) {
  size_t i = (size_t)blockIdx.x * blockDim.x + threadIdx.x;
  if (i < (size_t)NPTS * NFEAT) {
    size_t r = i >> 8, c = i & 255;
    xn[r * XNLD + c] = (_Float16)x[i];
  }
}

// ============================================================
// Kernel 0b: W_proj f32 -> f16 (same row-major layout)
// ============================================================
__global__ void k_prep_w(const float* __restrict__ wpf, _Float16* __restrict__ wp16) {
  int i = blockIdx.x * blockDim.x + threadIdx.x;
  if (i < NEMBED * 2 * NFEAT) wp16[i] = (_Float16)wpf[i];
}

// ============================================================
// Kernel 0c: XT = x^T as f16, [256][16384]. Tiled 32x32 transpose
// through LDS so both global access phases are coalesced.
// ============================================================
__global__ __launch_bounds__(256) void k_prep_xt(const float* __restrict__ x,
                                                 _Float16* __restrict__ xt) {
  __shared__ _Float16 t[32][33];
  const int tid = threadIdx.x;
  const int r0 = (blockIdx.x & (NPTS / 32 - 1)) * 32;   // 512 tiles along N
  const int c0 = (blockIdx.x / (NPTS / 32)) * 32;       // 8 tiles along feat
  #pragma unroll
  for (int i = 0; i < 4; ++i) {
    int idx = tid + i * 256;
    int rr = idx >> 5, cc = idx & 31;
    t[rr][cc] = (_Float16)x[(size_t)(r0 + rr) * NFEAT + (c0 + cc)];
  }
  __syncthreads();
  #pragma unroll
  for (int i = 0; i < 4; ++i) {
    int idx = tid + i * 256;
    int cc = idx >> 5, rr = idx & 31;
    xt[(size_t)(c0 + cc) * NPTS + (r0 + rr)] = t[rr][cc];
  }
}

// ============================================================
// Kernel 1: S = adj @ x (f16 WMMA, f32 acc), fused row-sum -> deg,
//           writes neigh = S/deg as f16 into XN[:, 256:512].
// Block: 512 threads (16 waves), 64 rows x 256 cols of output.
// adj tiles (f32) arrive via double-buffered tensor_load_to_lds;
// B fragments stream from pre-transposed XT with an explicit
// software pipeline (bfA/bfB) so L2 latency hides under the WMMAs.
// ============================================================
#define BM    64
#define KT    64
#define LDA32 68   // f32 At row stride in dwords (64 + 4 pad, set by TDM pad fields)

__global__ __launch_bounds__(512) void k_sage_gemm(
    const float* __restrict__ adj,
    const _Float16* __restrict__ xt,      // [256][16384] f16
    _Float16* __restrict__ xn)            // writes cols [256,512)
{
  __shared__ float At32[2][BM * LDA32] __attribute__((aligned(16)));
  __shared__ float degArr[BM];

  const int tid  = threadIdx.x;
  const int lane = tid & 31;
  const int w    = tid >> 5;   // wave 0..15
  const int rt   = w >> 2;     // row tile 0..3 (16 rows each)
  const int cg   = w & 3;      // col group: tiles 4*cg .. 4*cg+3
  const int cgu  = __builtin_amdgcn_readfirstlane(cg);  // scalar copy: keeps EXEC full
  const int lh   = lane >> 4;  // 0/1
  const int lm   = lane & 15;
  const int rowBase = blockIdx.x * BM;

  v8f acc[4];
  v8f rs;
  #pragma unroll
  for (int i = 0; i < 8; ++i) { rs[i] = 0.f; acc[0][i] = 0.f; acc[1][i] = 0.f; acc[2][i] = 0.f; acc[3][i] = 0.f; }

  // constant B fragment: column 0 all ones -> accumulator col 0 = row sums
  v16h bones;
  {
    _Float16 ov = (lm == 0) ? (_Float16)1.0f : (_Float16)0.0f;
    #pragma unroll
    for (int i = 0; i < 16; ++i) bones[i] = ov;
  }

  // per-wave B source columns in XT (per lane two 8-half K runs, 16 apart)
  const _Float16* bcol[4];
  #pragma unroll
  for (int j = 0; j < 4; ++j)
    bcol[j] = &xt[(size_t)((4 * cg + j) * 16 + lm) * NPTS + 8 * lh];

  const unsigned int atOff[2] = {
    (unsigned int)(size_t)&At32[0][0],
    (unsigned int)(size_t)&At32[1][0]
  };

  // A fragment from f32 LDS tile (convert at load)
  auto load_af = [&](int cur, int ks) -> v16h {
    const float* ap = &At32[cur][(rt * 16 + lm) * LDA32 + ks * 32 + 8 * lh];
    v16h af;
    #pragma unroll
    for (int i = 0; i < 8; ++i) af[i] = (_Float16)ap[i];
    #pragma unroll
    for (int i = 0; i < 8; ++i) af[8 + i] = (_Float16)ap[16 + i];
    return af;
  };
  // 4 B fragments for one 32-wide K step starting at kbase
  auto load_bf = [&](v16h* bf, int kbase) {
    #pragma unroll
    for (int j = 0; j < 4; ++j) {
      const _Float16* bb = bcol[j] + kbase;
      bf[j] = pack16(*(const v8h*)bb, *(const v8h*)(bb + 16));
    }
  };

  // ---- prologue: DMA first adj tile; preload first B fragments ----
  v16h bfA[4], bfB[4];
  load_bf(bfA, 0);
  if (tid < 32) {
    tdm_load_tile_f32(atOff[0], &adj[(size_t)rowBase * NPTS],
                      KT, BM, NPTS, NPTS, NPTS);
    wait_tensor0();
  }
  __syncthreads();

  for (int it = 0; it < NPTS / KT; ++it) {
    const int cur = it & 1, nxt = cur ^ 1;
    const int k0  = it * KT;
    const int k0n = k0 + KT;

    // ---- kick off next DMA while computing on current ----
    if (k0n < NPTS) {
      if (tid < 32)
        tdm_load_tile_f32(atOff[nxt], &adj[(size_t)rowBase * NPTS + k0n],
                          KT, BM, NPTS, NPTS, NPTS);
      if (k0n + KT < NPTS)
        __builtin_prefetch(&adj[(size_t)(rowBase + (tid >> 3)) * NPTS + k0n + KT + (tid & 7) * 8], 0, 1);
    }

    // ---- ks = 0: consume bfA, prefetch bfB (k0+32) ----
    {
      v16h af = load_af(cur, 0);
      load_bf(bfB, k0 + 32);                    // in flight under the burst below
      #pragma unroll
      for (int j = 0; j < 4; ++j) acc[j] = wmma_f16(af, bfA[j], acc[j]);
      if (cgu == 0) rs = wmma_f16(af, bones, rs);   // scalar branch, EXEC all-ones
    }
    // ---- ks = 1: consume bfB, prefetch bfA (next iteration) ----
    {
      v16h af = load_af(cur, 1);
      if (k0n < NPTS) load_bf(bfA, k0n);        // in flight under the burst below
      #pragma unroll
      for (int j = 0; j < 4; ++j) acc[j] = wmma_f16(af, bfB[j], acc[j]);
      if (cgu == 0) rs = wmma_f16(af, bones, rs);
    }

    if (tid < 32) wait_tensor0();   // next tile landed before everyone passes barrier
    __syncthreads();
  }

  // deg = rowsum + 1 ; rowsum lives in col 0 => lanes 0 and 16 of cg==0 waves
  if (cg == 0 && lm == 0) {
    #pragma unroll
    for (int p = 0; p < 8; ++p)
      degArr[rt * 16 + 8 * lh + p] = rs[p] + 1.0f;
  }
  __syncthreads();

  // neigh = acc/deg -> f16 into XN cols [256,512)
  #pragma unroll
  for (int j = 0; j < 4; ++j) {
    #pragma unroll
    for (int p = 0; p < 8; ++p) {
      int row = rt * 16 + 8 * lh + p;
      float v = acc[j][p] / degArr[row];
      xn[(size_t)(rowBase + row) * XNLD + NFEAT + (4 * cg + j) * 16 + lm] = (_Float16)v;
    }
  }
}

// ============================================================
// Kernel 2: h^T = Wp(f16) @ XN^T via WMMA (frags straight from global),
// then ReLU -> LDS -> 3-layer leaky-ReLU MLP -> softmax -> d_out.
// Block: 256 threads (8 waves) handles 16 graph rows; wave w = output
// channels [16w,16w+16).
// ============================================================
__global__ __launch_bounds__(256) void k_mlp(
    const _Float16* __restrict__ xn,    // [N][512] f16
    const _Float16* __restrict__ wp,    // [128][512] f16
    const float* __restrict__ W1, const float* __restrict__ B1,
    const float* __restrict__ W2, const float* __restrict__ B2,
    const float* __restrict__ W3, const float* __restrict__ B3,
    float* __restrict__ out)
{
  __shared__ float hb [16][NEMBED + 1];
  __shared__ float h1b[16][HID1 + 1];
  __shared__ float h2b[16][HID2 + 1];
  __shared__ float zb [16][NOUT + 1];

  const int tid = threadIdx.x, lane = tid & 31, w = tid >> 5;
  const int lh = lane >> 4, lm = lane & 15;
  const int rowBase = blockIdx.x * 16;

  v8f acc;
  #pragma unroll
  for (int i = 0; i < 8; ++i) acc[i] = 0.f;

  const _Float16* arow = &wp[(size_t)(w * 16 + lm) * XNLD];       // Wp channel row
  const _Float16* brow = &xn[(size_t)(rowBase + lm) * XNLD];      // XN graph row (as B column)
  #pragma unroll
  for (int ks = 0; ks < XNLD / 32; ++ks) {
    const _Float16* ab = arow + ks * 32 + 8 * lh;
    v16h af = pack16(*(const v8h*)ab, *(const v8h*)(ab + 16));
    const _Float16* bb = brow + ks * 32 + 8 * lh;
    v16h bf = pack16(*(const v8h*)bb, *(const v8h*)(bb + 16));
    acc = wmma_f16(af, bf, acc);
  }
  // C^T: row M = channel = 16w + p + 8*lh, col = graph row = lm
  #pragma unroll
  for (int p = 0; p < 8; ++p)
    hb[lm][w * 16 + 8 * lh + p] = fmaxf(acc[p], 0.0f);   // ReLU
  __syncthreads();

  // layer 1: 16x64, dot over 128
  for (int idx = tid; idx < 16 * HID1; idx += 256) {
    int j = idx >> 6, o = idx & (HID1 - 1);
    float s = B1[o];
    #pragma unroll 8
    for (int k = 0; k < NEMBED; ++k) s += hb[j][k] * W1[o * NEMBED + k];
    h1b[j][o] = (s > 0.f) ? s : 0.01f * s;
  }
  __syncthreads();
  // layer 2: 16x32, dot over 64
  for (int idx = tid; idx < 16 * HID2; idx += 256) {
    int j = idx >> 5, o = idx & (HID2 - 1);
    float s = B2[o];
    #pragma unroll 8
    for (int k = 0; k < HID1; ++k) s += h1b[j][k] * W2[o * HID1 + k];
    h2b[j][o] = (s > 0.f) ? s : 0.01f * s;
  }
  __syncthreads();
  // layer 3: 16x8, dot over 32
  if (tid < 16 * NOUT) {
    int j = tid >> 3, o = tid & 7;
    float s = B3[o];
    #pragma unroll
    for (int k = 0; k < HID2; ++k) s += h2b[j][k] * W3[o * HID2 + k];
    zb[j][o] = (s > 0.f) ? s : 0.01f * s;
  }
  __syncthreads();
  // softmax over 8 outputs per row
  if (tid < 16) {
    int j = tid;
    float m = zb[j][0];
    #pragma unroll
    for (int o = 1; o < 8; ++o) m = fmaxf(m, zb[j][o]);
    float e[8], sum = 0.f;
    #pragma unroll
    for (int o = 0; o < 8; ++o) { e[o] = expf(zb[j][o] - m); sum += e[o]; }
    float inv = 1.f / sum;
    #pragma unroll
    for (int o = 0; o < 8; ++o) out[(size_t)(rowBase + j) * NOUT + o] = e[o] * inv;
  }
}

// ============================================================
extern "C" void kernel_launch(void* const* d_in, const int* in_sizes, int n_in,
                              void* d_out, int out_size, void* d_ws, size_t ws_size,
                              hipStream_t stream) {
  const float* x   = (const float*)d_in[0];
  const float* adj = (const float*)d_in[1];
  const float* Wp  = (const float*)d_in[2];
  const float* W1  = (const float*)d_in[3];
  const float* B1  = (const float*)d_in[4];
  const float* W2  = (const float*)d_in[5];
  const float* B2  = (const float*)d_in[6];
  const float* W3  = (const float*)d_in[7];
  const float* B3  = (const float*)d_in[8];

  _Float16* XN   = (_Float16*)d_ws;                                     // 16 MB: [N][512] f16
  _Float16* WP16 = (_Float16*)((char*)d_ws +
                               (size_t)NPTS * XNLD * sizeof(_Float16)); // 128 KB
  _Float16* XT   = (_Float16*)((char*)d_ws +
                               (size_t)NPTS * XNLD * sizeof(_Float16) +
                               (size_t)NEMBED * 2 * NFEAT * sizeof(_Float16)); // 8 MB

  k_prep_x<<<(NPTS * NFEAT) / 256, 256, 0, stream>>>(x, XN);
  k_prep_w<<<(NEMBED * 2 * NFEAT) / 256, 256, 0, stream>>>(Wp, WP16);
  k_prep_xt<<<(NPTS / 32) * (NFEAT / 32), 256, 0, stream>>>(x, XT);
  k_sage_gemm<<<NPTS / BM, 512, 0, stream>>>(adj, XT, XN);
  k_mlp<<<NPTS / 16, 256, 0, stream>>>(XN, WP16, W1, B1, W2, B2, W3, B3,
                                       (float*)d_out);
}
